// DeepMatchingLoss_43954695307636
// MI455X (gfx1250) — matrix operand order repaired and checked by reading, not compiled, not run on hardware
//
#include <hip/hip_runtime.h>
#include <hip/hip_bf16.h>
#include <math.h>

typedef __attribute__((ext_vector_type(16))) __bf16      v16bf;
typedef __attribute__((ext_vector_type(8)))  float       v8f;
typedef __attribute__((ext_vector_type(8)))  unsigned int v8u;

union Frag { v8u u; v16bf b; };

#define BATCH   4
#define DDIM    128
#define IMG     256
#define MTILES  49      // 784 / 16
#define NTILES  784     // 12544 / 16
#define KBLK    4       // 128 / 32
#define NQ_DW   (BATCH*MTILES*KBLK*32*8)   // 200704 dwords
#define NT_DW   (BATCH*NTILES*KBLK*32*8)   // 3211264 dwords

__device__ __forceinline__ unsigned short f2bf(float f){
  unsigned u = __float_as_uint(f);
  unsigned r = u + 0x7FFFu + ((u >> 16) & 1u);   // round-to-nearest-even
  return (unsigned short)(r >> 16);
}
__device__ __forceinline__ float bf2f(unsigned short h){
  return __uint_as_float(((unsigned)h) << 16);
}

// ---------------------------------------------------------------------------
// Pack one descriptor tensor (B,128,256,256 f32) into WMMA fragment arrays:
//   Q (queries,  rows/cols 20+8j, 28x28)  -> A-matrix 16-bit layout
//   T (targets,  rows/cols 16+2i, 112x112)-> B-matrix 16-bit layout
// Each f32 is split x = hi(bf16) + lo(bf16) for a 3-product near-fp32 GEMM.
// ---------------------------------------------------------------------------
__global__ void dm_pack_kernel(const float* __restrict__ desc,
                               unsigned* __restrict__ Qhi, unsigned* __restrict__ Qlo,
                               unsigned* __restrict__ Thi, unsigned* __restrict__ Tlo){
  int idx = blockIdx.x * blockDim.x + threadIdx.x;
  if (idx < NQ_DW){
    int i = idx;
    int d    = i & 7;  i >>= 3;
    int lane = i & 31; i >>= 5;
    int kb   = i & 3;  i >>= 2;
    int tile = i % MTILES;
    int b    = i / MTILES;
    int mloc = lane & 15;
    int M    = tile * 16 + mloc;
    int h1   = M / 28, w1 = M % 28;
    int row  = 20 + 8*h1, col = 20 + 8*w1;
    // 16-bit A-matrix 16x32 layout: lanes<16 hold K {0..7,16..23}, lanes>=16 {8..15,24..31}
    int kh   = (lane >> 4) * 8;
    int kin  = (d < 4) ? (kh + 2*d) : (16 + kh + 2*(d - 4));
    int K    = kb * 32 + kin;
    size_t p0 = (((size_t)b*DDIM + K    )*IMG + row)*IMG + col;
    size_t p1 = (((size_t)b*DDIM + K + 1)*IMG + row)*IMG + col;
    float x0 = desc[p0], x1 = desc[p1];
    unsigned short bh0 = f2bf(x0), bh1 = f2bf(x1);
    unsigned short bl0 = f2bf(x0 - bf2f(bh0)), bl1 = f2bf(x1 - bf2f(bh1));
    Qhi[idx] = (unsigned)bh0 | ((unsigned)bh1 << 16);
    Qlo[idx] = (unsigned)bl0 | ((unsigned)bl1 << 16);
  } else if (idx < NQ_DW + NT_DW){
    int i = idx - NQ_DW;
    int o = i;
    int d    = i & 7;  i >>= 3;
    int lane = i & 31; i >>= 5;
    int kb   = i & 3;  i >>= 2;
    int tile = i % NTILES;
    int b    = i / NTILES;
    int nloc = lane & 15;
    int N    = tile * 16 + nloc;
    int y    = N / 112, x = N % 112;
    int row  = 16 + 2*y, col = 16 + 2*x;
    // 16-bit B-matrix 32x16 layout: lanes<16 hold K {0..7,16..23}, lanes>=16 {8..15,24..31}
    int K    = kb * 32 + ((d < 4) ? ((lane >> 4) * 8 + 2*d) : (16 + (lane >> 4) * 8 + 2*(d - 4)));
    size_t p0 = (((size_t)b*DDIM + K    )*IMG + row)*IMG + col;
    size_t p1 = (((size_t)b*DDIM + K + 1)*IMG + row)*IMG + col;
    float x0 = desc[p0], x1 = desc[p1];
    unsigned short bh0 = f2bf(x0), bh1 = f2bf(x1);
    unsigned short bl0 = f2bf(x0 - bf2f(bh0)), bl1 = f2bf(x1 - bf2f(bh1));
    Thi[o] = (unsigned)bh0 | ((unsigned)bh1 << 16);
    Tlo[o] = (unsigned)bl0 | ((unsigned)bl1 << 16);
  }
}

// ---------------------------------------------------------------------------
// Correlation GEMM: corr[b, m(784), n(12544)] = sum_k Q[m,k] * T[n,k]
// One wave -> one 16(M) x 64(N) tile: A fragments reused across 4 N-tiles,
// 3 WMMAs per k-block (hi*hi + hi*lo + lo*hi) ~ fp32 accuracy.
// ---------------------------------------------------------------------------
__global__ void __launch_bounds__(256)
dm_gemm_kernel(const v8u* __restrict__ Ahi, const v8u* __restrict__ Alo,
               const v8u* __restrict__ Bhi, const v8u* __restrict__ Blo,
               float* __restrict__ corr){
  int wid = (blockIdx.x * blockDim.x + threadIdx.x) >> 5;
  if (wid >= BATCH * MTILES * 196) return;   // uniform per wave
  int lane   = threadIdx.x & 31;
  int ngroup = wid % 196;
  int rest   = wid / 196;
  int mtile  = rest % MTILES;
  int b      = rest / MTILES;

  Frag ahi[KBLK], alo[KBLK];
  size_t baseA = ((size_t)(b * MTILES + mtile) * KBLK) * 32 + lane;
#pragma unroll
  for (int kb = 0; kb < KBLK; ++kb){
    ahi[kb].u = Ahi[baseA + kb * 32];
    alo[kb].u = Alo[baseA + kb * 32];
  }

  int nn = lane & 15;
  int mh = (lane >> 4) * 8;
#pragma unroll
  for (int nt = 0; nt < 4; ++nt){
    int ntile = ngroup * 4 + nt;
    size_t baseB = ((size_t)(b * NTILES + ntile) * KBLK) * 32 + lane;
    if (nt < 3){ // prefetch next N-tile's B fragments (global_prefetch_b8)
      __builtin_prefetch((const void*)&Bhi[baseB + 128], 0, 1);
      __builtin_prefetch((const void*)&Blo[baseB + 128], 0, 1);
    }
    v8f acc = {0.f,0.f,0.f,0.f,0.f,0.f,0.f,0.f};
#pragma unroll
    for (int kb = 0; kb < KBLK; ++kb){
      Frag bh, bl;
      bh.u = Bhi[baseB + kb * 32];
      bl.u = Blo[baseB + kb * 32];
      acc = __builtin_amdgcn_wmma_f32_16x16x32_bf16(false, alo[kb].b, false, bh.b, (short)0, acc, false, false);
      acc = __builtin_amdgcn_wmma_f32_16x16x32_bf16(false, ahi[kb].b, false, bl.b, (short)0, acc, false, false);
      acc = __builtin_amdgcn_wmma_f32_16x16x32_bf16(false, ahi[kb].b, false, bh.b, (short)0, acc, false, false);
    }
    // C/D layout: VGPR r holds M = mh + r, lane&15 holds N
    size_t ncol = (size_t)ntile * 16 + nn;
#pragma unroll
    for (int r = 0; r < 8; ++r){
      int m = mtile * 16 + mh + r;
      corr[((size_t)b * 784 + m) * 12544 + ncol] = acc[r];
    }
  }
}

// ---------------------------------------------------------------------------
// One pyramid level, fused: 3x3/s2/pad1 maxpool + relu^1.5 + shifted 4-child
// aggregation + count^-0.9 normalization.  in: (B,H1,W1,H2,W2)  out: (B,Hp,Wp,H2p,W2p)
// ---------------------------------------------------------------------------
__global__ void dm_level_kernel(const float* __restrict__ in, float* __restrict__ out,
                                int H1, int W1, int H2, int W2,
                                int Hp, int Wp, int H2p, int W2p, int total){
  int i = blockIdx.x * blockDim.x + threadIdx.x;
  if (i >= total) return;
  int idx = i;
  int v  = idx % W2p; idx /= W2p;
  int u  = idx % H2p; idx /= H2p;
  int wp = idx % Wp;  idx /= Wp;
  int hp = idx % Hp;
  int b  = idx / Hp;

  float s = 0.f;
#pragma unroll
  for (int dy = 0; dy < 2; ++dy){
    int up = u + (dy ? 1 : -1);
    if ((unsigned)up >= (unsigned)H2p) continue;
    int h1 = 2 * hp + dy;
#pragma unroll
    for (int dx = 0; dx < 2; ++dx){
      int vp = v + (dx ? 1 : -1);
      if ((unsigned)vp >= (unsigned)W2p) continue;
      int w1 = 2 * wp + dx;
      const float* base = in + ((size_t)((b * H1 + h1) * W1 + w1) * H2) * W2;
      int y0 = 2 * up - 1, y1 = 2 * up + 1;
      if (y0 < 0) y0 = 0;
      if (y1 > H2 - 1) y1 = H2 - 1;
      int x0 = 2 * vp - 1, x1 = 2 * vp + 1;
      if (x0 < 0) x0 = 0;
      if (x1 > W2 - 1) x1 = W2 - 1;
      float m = -__builtin_inff();
      for (int yy = y0; yy <= y1; ++yy)
        for (int xx = x0; xx <= x1; ++xx)
          m = fmaxf(m, base[(size_t)yy * W2 + xx]);
      float r = fmaxf(m, 0.f);
      s += r * sqrtf(r);                       // relu(x)^1.5
    }
  }
  float rc = (u == 0 || u == H2p - 1) ? 1.f : 2.f;
  float cc = (v == 0 || v == W2p - 1) ? 1.f : 2.f;
  float cnt = rc * cc;                          // 1, 2 or 4
  float inv = (cnt == 1.f) ? 1.f : ((cnt == 2.f) ? 0.53588673f   // 2^-0.9
                                                 : 0.28717459f); // 4^-0.9
  out[i] = s * inv;
}

// Final reduction of (4,1,1,7,7): rect, normalize, -mean of sum of squares.
// Single-threaded & stream-ordered -> deterministic.
__global__ void dm_final_kernel(const float* __restrict__ c, float* __restrict__ slot){
  if (threadIdx.x == 0 && blockIdx.x == 0){
    float total = 0.f;
    for (int b = 0; b < BATCH; ++b){
      float denom = 0.f;
      for (int i = 0; i < 49; ++i){
        float x = fmaxf(c[b * 49 + i], 0.f);
        denom += x * sqrtf(x);
      }
      denom += 0.03f;                           // EPS
      float s = 0.f;
      for (int i = 0; i < 49; ++i){
        float x = fmaxf(c[b * 49 + i], 0.f);
        float q = (x * sqrtf(x)) / denom;
        s += q * q;
      }
      total += s;
    }
    slot[0] = -total / (float)BATCH;
  }
}

__global__ void dm_combine_kernel(const float* __restrict__ slots, float* __restrict__ out){
  if (threadIdx.x == 0 && blockIdx.x == 0)
    out[0] = 0.5f * (slots[0] + slots[1]);
}

// ---------------------------------------------------------------------------
extern "C" void kernel_launch(void* const* d_in, const int* in_sizes, int n_in,
                              void* d_out, int out_size, void* d_ws, size_t ws_size,
                              hipStream_t stream){
  (void)in_sizes; (void)n_in; (void)out_size; (void)ws_size;
  const float* desc1 = (const float*)d_in[0];
  const float* desc2 = (const float*)d_in[1];
  float* out = (float*)d_out;

  char* ws = (char*)d_ws;
  size_t off = 0;
  auto take = [&](size_t bytes) -> char* {
    char* p = ws + off;
    off += (bytes + 255) & ~(size_t)255;
    return p;
  };
  const size_t Qb = (size_t)NQ_DW * 4;
  const size_t Tb = (size_t)NT_DW * 4;
  unsigned* Q1h = (unsigned*)take(Qb); unsigned* Q1l = (unsigned*)take(Qb);
  unsigned* T1h = (unsigned*)take(Tb); unsigned* T1l = (unsigned*)take(Tb);
  unsigned* Q2h = (unsigned*)take(Qb); unsigned* Q2l = (unsigned*)take(Qb);
  unsigned* T2h = (unsigned*)take(Tb); unsigned* T2l = (unsigned*)take(Tb);
  float* c0 = (float*)take((size_t)BATCH * 784 * 12544 * 4);   // (4,28,28,112,112)
  float* c1 = (float*)take((size_t)BATCH * 196 * 3136 * 4);    // (4,14,14,56,56)
  float* c2 = (float*)take((size_t)BATCH * 49 * 784 * 4);      // (4,7,7,28,28)
  float* c3 = (float*)take((size_t)BATCH * 9 * 196 * 4);       // (4,3,3,14,14)
  float* c4 = (float*)take((size_t)BATCH * 49 * 4);            // (4,1,1,7,7)
  float* slots = (float*)take(2 * sizeof(float));

  const int packThreads = NQ_DW + NT_DW;
  const int packBlocks  = (packThreads + 255) / 256;
  dm_pack_kernel<<<packBlocks, 256, 0, stream>>>(desc1, Q1h, Q1l, T1h, T1l);
  dm_pack_kernel<<<packBlocks, 256, 0, stream>>>(desc2, Q2h, Q2l, T2h, T2l);

  const int gemmWaves  = BATCH * MTILES * 196;                 // 38416
  const int gemmBlocks = (gemmWaves * 32 + 255) / 256;         // 4802

  for (int dir = 0; dir < 2; ++dir){
    const unsigned* Ah = dir ? Q2h : Q1h;
    const unsigned* Al = dir ? Q2l : Q1l;
    const unsigned* Bh = dir ? T1h : T2h;
    const unsigned* Bl = dir ? T1l : T2l;
    dm_gemm_kernel<<<gemmBlocks, 256, 0, stream>>>(
        (const v8u*)Ah, (const v8u*)Al, (const v8u*)Bh, (const v8u*)Bl, c0);

    int t0 = BATCH * 14 * 14 * 56 * 56;
    dm_level_kernel<<<(t0 + 255) / 256, 256, 0, stream>>>(c0, c1, 28, 28, 112, 112, 14, 14, 56, 56, t0);
    int t1 = BATCH * 7 * 7 * 28 * 28;
    dm_level_kernel<<<(t1 + 255) / 256, 256, 0, stream>>>(c1, c2, 14, 14, 56, 56, 7, 7, 28, 28, t1);
    int t2 = BATCH * 3 * 3 * 14 * 14;
    dm_level_kernel<<<(t2 + 255) / 256, 256, 0, stream>>>(c2, c3, 7, 7, 28, 28, 3, 3, 14, 14, t2);
    int t3 = BATCH * 1 * 1 * 7 * 7;
    dm_level_kernel<<<1, 256, 0, stream>>>(c3, c4, 3, 3, 14, 14, 1, 1, 7, 7, t3);

    dm_final_kernel<<<1, 32, 0, stream>>>(c4, slots + dir);
  }
  dm_combine_kernel<<<1, 32, 0, stream>>>(slots, out);
}
